// UserAttentionPooling_6313601925681
// MI455X (gfx1250) — matrix operand order repaired and checked by reading, not compiled, not run on hardware
//
#include <hip/hip_runtime.h>
#include <hip/hip_bf16.h>

typedef __attribute__((ext_vector_type(16))) _Float16     v16h;
typedef __attribute__((ext_vector_type(8)))  float        v8f;
typedef __attribute__((ext_vector_type(4)))  float        f4;
typedef __attribute__((ext_vector_type(4)))  unsigned int u32x4;
typedef __attribute__((ext_vector_type(8)))  int          i32x8;
typedef __attribute__((ext_vector_type(4)))  int          i32x4;

#define DIM        128
#define CHUNK      16
#define LDS_STRIDE 132   // floats per staged row (128 + 4 pad -> conflict-free banks)

#define AS1 __attribute__((address_space(1)))
#define AS3 __attribute__((address_space(3)))

// ---- select the best CDNA5 staging path available on this toolchain ----
#if __has_builtin(__builtin_amdgcn_tensor_load_to_lds) && __has_builtin(__builtin_amdgcn_s_wait_tensorcnt)
#define STAGE_TDM 1
#elif __has_builtin(__builtin_amdgcn_global_load_async_to_lds_b128) && __has_builtin(__builtin_amdgcn_s_wait_asynccnt)
#define STAGE_ASYNC 1
#endif

#if defined(STAGE_TDM)
// One TDM descriptor DMAs a 16x128 f32 tile global->LDS with the 4-DWORD row
// padding applied by the engine (pad_interval=128 DW -> code 6, pad_amount=4 DW
// -> code 3). tensor_dim1 = rows remaining in the segment so the tail tile is
// zero-filled by the OOB rule; those rows are masked out of the softmax anyway.
__device__ __forceinline__ void stage_tdm(const float* E, int row0, int seg_end,
                                          float* buf) {
    const unsigned int rows_rem = (unsigned int)(seg_end - row0);      // >= 1
    const unsigned long long gaddr =
        (unsigned long long)(uintptr_t)(E + (size_t)row0 * DIM);
    const unsigned int lds = (unsigned int)(uintptr_t)buf;             // addr[31:0] = LDS byte addr

    u32x4 g0;
    g0.x = 1u;                                                          // count=1 (valid user D#)
    g0.y = lds;                                                         // lds_addr
    g0.z = (unsigned int)gaddr;                                         // global_addr[31:0]
    g0.w = (unsigned int)((gaddr >> 32) & 0x01FFFFFFu) | (2u << 30);    // global_addr[56:32] | type=2

    i32x8 g1;
    g1[0] = (2 << 16) | (1 << 20) | (6 << 22) | (3 << 25);  // data_size=4B, pad_enable, interval=128DW, amount=4DW
    g1[1] = (int)(((unsigned)DIM & 0xFFFFu) << 16);         // tensor_dim0[15:0] = 128
    g1[2] = (int)((rows_rem & 0xFFFFu) << 16);              // tensor_dim1[15:0]
    g1[3] = (int)(((rows_rem >> 16) & 0xFFFFu) | ((unsigned)DIM << 16)); // tensor_dim1[31:16] | tile_dim0=128
    g1[4] = CHUNK;                                          // tile_dim1=16, tile_dim2=0
    g1[5] = DIM;                                            // tensor_dim0_stride = 128 (low 32)
    g1[6] = 0;                                              // stride hi / dim1_stride lo
    g1[7] = 0;

    const i32x4 gz4 = {0, 0, 0, 0};
#if defined(__clang_major__) && (__clang_major__ >= 23)
    const i32x8 gz8 = {0, 0, 0, 0, 0, 0, 0, 0};
    __builtin_amdgcn_tensor_load_to_lds(g0, g1, gz4, gz4, gz8, 0);
#else
    __builtin_amdgcn_tensor_load_to_lds(g0, g1, gz4, gz4, 0);
#endif
}
#elif defined(STAGE_ASYNC)
// 16 x global_load_async_to_lds_b128: each op moves one 512B row (32 lanes x 16B)
// with per-lane LDS addresses implementing the padded 132-float stride.
// Tail rows clamp to the last valid row (finite data; weight is 0 later).
__device__ __forceinline__ void stage_async(const float* E, int row0, int seg_end,
                                            float* buf, int lane) {
#pragma unroll
    for (int r = 0; r < CHUNK; ++r) {
        int row = row0 + r;
        if (row >= seg_end) row = seg_end - 1;
        const float* src = E + (size_t)row * DIM + 4 * lane;
        float*       dst = buf + r * LDS_STRIDE + 4 * lane;
        __builtin_amdgcn_global_load_async_to_lds_b128((AS1 void*)src, (AS3 void*)dst, 0, 0);
    }
}
#endif

// One wave32 workgroup per user segment. Flash-style single pass:
// DMA/stage 16 rows to LDS (double-buffered), scores via v_wmma_f32_16x16x32_f16
// (q in column 0 of B), online-softmax rescale, VALU weighted accumulate.
__global__ __launch_bounds__(32) void UserAttentionPooling_kernel(
    const float* __restrict__ E,      // [T, 128]
    const float* __restrict__ q,      // [128]
    const int*   __restrict__ offs,   // [U+1]
    float*       __restrict__ out)    // [U, 128]
{
    __shared__ float tile[2][CHUNK * LDS_STRIDE];
    __shared__ float q_s[DIM];
    __shared__ float sc_s[CHUNK];

    const int u     = blockIdx.x;
    const int lane  = threadIdx.x;         // 0..31
    const int start = offs[u];
    const int end   = offs[u + 1];
    const int len   = end - start;
    const int nchunks = (len + CHUNK - 1) / CHUNK;

    // ---- stage q into LDS (one b128 per lane) ----
    *(f4*)(q_s + 4 * lane) = *(const f4*)(q + 4 * lane);
    __syncthreads();

    // ---- build the 4 loop-invariant B fragments: B[k,0] = q[k0+k], else 0 ----
    // 16-bit B (32x16): lanes 0-15 hold K = {0..7, 16..23} (col=lane),
    // lanes 16-31 hold K = {8..15, 24..31} (col=lane-16). Column 0 lives on lanes 0 & 16.
    v16h bfrag[4];
#pragma unroll
    for (int kb = 0; kb < 4; ++kb) {
        v16h b;
#pragma unroll
        for (int i = 0; i < 16; ++i) b[i] = (_Float16)0.0f;
        if (lane == 0 || lane == 16) {
            const int col0 = kb * 32 + ((lane == 16) ? 8 : 0);
#pragma unroll
            for (int i = 0; i < 16; ++i) b[i] = (_Float16)q_s[col0 + (i < 8 ? i : 8 + i)];
        }
        bfrag[kb] = b;
    }

    // ---- online softmax state ----
    float m = -INFINITY;
    float s = 0.0f;
    f4 acc = {0.0f, 0.0f, 0.0f, 0.0f};    // dims [4*lane, 4*lane+3]

#if defined(STAGE_TDM)
    stage_tdm(E, start, end, tile[0]);                 // prologue: chunk 0 in flight
#elif defined(STAGE_ASYNC)
    stage_async(E, start, end, tile[0], lane);
#endif

    for (int ci = 0; ci < nchunks; ++ci) {
        const int base   = ci * CHUNK;
        const int nvalid = (len - base < CHUNK) ? (len - base) : CHUNK;
        float* tbuf = tile[ci & 1];

#if defined(STAGE_TDM)
        if (ci + 1 < nchunks) {
            stage_tdm(E, start + base + CHUNK, end, tile[(ci + 1) & 1]);
            __builtin_amdgcn_s_wait_tensorcnt(1);      // chunk ci landed, ci+1 in flight
        } else {
            __builtin_amdgcn_s_wait_tensorcnt(0);
        }
#elif defined(STAGE_ASYNC)
        if (ci + 1 < nchunks) {
            stage_async(E, start + base + CHUNK, end, tile[(ci + 1) & 1], lane);
            __builtin_amdgcn_s_wait_asynccnt(16);      // first 16 (chunk ci) complete in order
        } else {
            __builtin_amdgcn_s_wait_asynccnt(0);
        }
#else
        // fallback: coalesced b128 loads + ds_store, zero-padded tail
#pragma unroll
        for (int r = 0; r < CHUNK; ++r) {
            f4 v = {0.0f, 0.0f, 0.0f, 0.0f};
            if (r < nvalid)
                v = *(const f4*)(E + (size_t)(start + base + r) * DIM + 4 * lane);
            *(f4*)(tbuf + r * LDS_STRIDE + 4 * lane) = v;
        }
        if (base + CHUNK < len) {       // prefetch next chunk (32 x 256B granules)
            const size_t prow = (size_t)(start + base + CHUNK + (lane >> 1)) * DIM
                              + (size_t)(lane & 1) * 64;
            __builtin_prefetch(E + prow, 0, 1);
        }
#endif
        __syncthreads();

        // ---- scores[16] = E_tile(16x128) * q(128) via 4 chained WMMA f16 ----
        // 16-bit A (16x32): lane<16 -> token=lane, K = {k0..k0+7, k0+16..k0+23};
        //                   lane>=16 -> token=lane-16, K = {k0+8..k0+15, k0+24..k0+31}.
        v8f c = {0.0f, 0.0f, 0.0f, 0.0f, 0.0f, 0.0f, 0.0f, 0.0f};
#pragma unroll
        for (int kb = 0; kb < 4; ++kb) {
            const int t     = lane & 15;
            const int kbase = kb * 32 + ((lane & 16) ? 8 : 0);
            const float* rowp = tbuf + t * LDS_STRIDE + kbase;
            v16h a;
#pragma unroll
            for (int i = 0; i < 8; ++i) a[i]     = (_Float16)rowp[i];
#pragma unroll
            for (int i = 0; i < 8; ++i) a[8 + i] = (_Float16)rowp[16 + i];
            c = __builtin_amdgcn_wmma_f32_16x16x32_f16(
                    false, a, false, bfrag[kb], (short)0, c, false, false);
        }

        // ---- broadcast column 0 of D: lane 0 holds M=0..7, lane 16 holds M=8..15 ----
        if ((lane & 15) == 0) {
            const int mbase = lane >> 1;   // lane0 -> 0, lane16 -> 8
#pragma unroll
            for (int i = 0; i < 8; ++i) sc_s[mbase + i] = c[i];
        }
        __syncthreads();

        // ---- online softmax update (redundant across lanes) ----
        float sc[CHUNK];
        float cmax = -INFINITY;
#pragma unroll
        for (int t = 0; t < CHUNK; ++t) {
            sc[t] = (t < nvalid) ? sc_s[t] : -INFINITY;
            cmax  = fmaxf(cmax, sc[t]);
        }
        const float m_new = fmaxf(m, cmax);
        const float r     = __expf(m - m_new);   // exp(-inf)=0 on first chunk
        s   *= r;
        acc *= r;

        float w[CHUNK];
#pragma unroll
        for (int t = 0; t < CHUNK; ++t) {
            w[t] = (t < nvalid) ? __expf(sc[t] - m_new) : 0.0f;
            s += w[t];
        }
        m = m_new;

        // ---- weighted accumulate: acc[d] += w_t * E_tile[t][d] ----
#pragma unroll
        for (int t = 0; t < CHUNK; ++t) {
            const f4 e = *(const f4*)(tbuf + t * LDS_STRIDE + 4 * lane);
            acc += e * w[t];
        }
        __syncthreads();   // tile buffer reused two chunks from now
    }

    const float inv = 1.0f / s;
    *(f4*)(out + (size_t)u * DIM + 4 * lane) = acc * inv;
}

extern "C" void kernel_launch(void* const* d_in, const int* in_sizes, int n_in,
                              void* d_out, int out_size, void* d_ws, size_t ws_size,
                              hipStream_t stream) {
    const float* E    = (const float*)d_in[0];   // his_embs [T,128]
    const float* q    = (const float*)d_in[1];   // query_vector [128]
    const int*   offs = (const int*)d_in[2];     // user_indices [U+1]
    float*       out  = (float*)d_out;           // [U,128]
    const int U = in_sizes[2] - 1;
    UserAttentionPooling_kernel<<<dim3(U), dim3(32), 0, stream>>>(E, q, offs, out);
}
